// MultiKmeansQuantizer_68384469287198
// MI455X (gfx1250) — matrix operand order repaired and checked by reading, not compiled, not run on hardware
//
#include <hip/hip_runtime.h>
#include <hip/hip_bf16.h>
#include <math.h>

// Problem constants (from reference)
#define BQ    16384        // frames
#define DIMQ  512          // feature dim
#define CQ    8            // codebooks
#define KQ    256          // codebook size
#define NFLAT (CQ * KQ)    // 2048 flattened centers
#define EPSQ  1e-20f

typedef __attribute__((ext_vector_type(16))) __bf16 bf16x16;
typedef __attribute__((ext_vector_type(8)))  __bf16 bf16x8;
typedef __attribute__((ext_vector_type(8)))  float  f32x8;

// ---------------- wave32 reduction helpers ----------------
__device__ __forceinline__ float wave_sum(float v) {
  #pragma unroll
  for (int off = 16; off >= 1; off >>= 1) v += __shfl_xor(v, off, 32);
  return v;
}
__device__ __forceinline__ float wave_max(float v) {
  #pragma unroll
  for (int off = 16; off >= 1; off >>= 1) v = fmaxf(v, __shfl_xor(v, off, 32));
  return v;
}
__device__ __forceinline__ unsigned int hash_u32(unsigned int a) {
  a ^= a >> 16; a *= 0x7feb352du; a ^= a >> 15; a *= 0x846ca68bu; a ^= a >> 16;
  return a;
}

// ---------------- init: zero accumulators, seed indexes ----------------
__global__ void init_kernel(const int* __restrict__ init_idx, int* __restrict__ idx,
                            float* __restrict__ avg_probs, float* __restrict__ scal) {
  int i = blockIdx.x * blockDim.x + threadIdx.x;
  if (i < BQ * CQ) idx[i] = init_idx[i];
  if (i < NFLAT)   avg_probs[i] = 0.0f;
  if (i < 4)       scal[i] = 0.0f;
}

// ---------------- fp32 -> bf16 conversion ----------------
__global__ void cvt_kernel(const float* __restrict__ in, __bf16* __restrict__ out, int n) {
  int i = blockIdx.x * blockDim.x + threadIdx.x;
  if (i < n) out[i] = (__bf16)in[i];
}

// ---------------- per-row squared norms of centers (wave per row) ----------------
__global__ __launch_bounds__(256)
void censq_kernel(const float* __restrict__ centers, float* __restrict__ censq) {
  int wave = threadIdx.x >> 5, lane = threadIdx.x & 31;
  int row  = blockIdx.x * 8 + wave;                  // NFLAT rows
  const float* p = centers + (size_t)row * DIMQ;
  float s = 0.0f;
  for (int d = lane; d < DIMQ; d += 32) { float v = p[d]; s += v * v; }
  s = wave_sum(s);
  if (lane == 0) censq[row] = s;
}

// ---------------- sum(x*x) global reduction ----------------
__global__ void xsq_kernel(const float* __restrict__ x, float* __restrict__ scal, int n) {
  float s = 0.0f;
  for (int i = blockIdx.x * blockDim.x + threadIdx.x; i < n; i += gridDim.x * blockDim.x) {
    float v = x[i]; s += v * v;
  }
  s = wave_sum(s);
  if ((threadIdx.x & 31) == 0) atomicAdd(&scal[0], s);
}

// ---------------- bf16 WMMA GEMM:  D[M,N] = A[M,Kd] * Bm[N,Kd]^T  ----------------
// 256 threads = 8 waves; wave tile 64(M) x 64(N) -> 16 WMMAs per 32-wide k-step,
// operand reuse 4x4, 16 b128 loads : 16 WMMAs. Block tile 128 x 256.
__global__ __launch_bounds__(256)
void gemm_bf16_nt(const __bf16* __restrict__ A, const __bf16* __restrict__ Bm,
                  float* __restrict__ D, int M, int N, int Kd) {
  const int lane = threadIdx.x & 31;
  const int wave = threadIdx.x >> 5;
  const int wm = wave & 1, wn = wave >> 1;           // 2 waves along M, 4 along N
  const int mBase = blockIdx.x * 128 + wm * 64;
  const int nBase = blockIdx.y * 256 + wn * 64;
  const int l15  = lane & 15;
  const int aoff = (lane >> 4) * 8;    // A: lanes 0-15 hold K {0..7,16..23}; 16-31 hold {8..15,24..31}
  const int boff = (lane >> 4) * 8;    // B: lanes 0-15 hold K {0..7,16..23}; 16-31 hold {8..15,24..31}

  f32x8 acc[4][4];
  #pragma unroll
  for (int i = 0; i < 4; ++i)
    #pragma unroll
    for (int j = 0; j < 4; ++j)
      acc[i][j] = (f32x8){0.f,0.f,0.f,0.f,0.f,0.f,0.f,0.f};

  for (int kk = 0; kk < Kd; kk += 32) {
    bf16x16 afrag[4], bfrag[4];
    #pragma unroll
    for (int i = 0; i < 4; ++i) {
      const __bf16* p = A + (size_t)(mBase + i * 16 + l15) * Kd + kk + aoff;
      bf16x8 lo = *(const bf16x8*)p;
      bf16x8 hi = *(const bf16x8*)(p + 16);
      #pragma unroll
      for (int t = 0; t < 8; ++t) { afrag[i][t] = lo[t]; afrag[i][t + 8] = hi[t]; }
    }
    #pragma unroll
    for (int j = 0; j < 4; ++j) {
      const __bf16* p = Bm + (size_t)(nBase + j * 16 + l15) * Kd + kk + boff;
      bf16x8 lo = *(const bf16x8*)p;
      bf16x8 hi = *(const bf16x8*)(p + 16);
      #pragma unroll
      for (int t = 0; t < 8; ++t) { bfrag[j][t] = lo[t]; bfrag[j][t + 8] = hi[t]; }
    }
    #pragma unroll
    for (int i = 0; i < 4; ++i)
      #pragma unroll
      for (int j = 0; j < 4; ++j)
        acc[i][j] = __builtin_amdgcn_wmma_f32_16x16x32_bf16(
            false, afrag[i], false, bfrag[j], (short)0, acc[i][j], false, false);
  }

  const int rowAdd = (lane >= 16) ? 8 : 0;   // C/D layout: lanes 16-31 carry rows M+8
  #pragma unroll
  for (int i = 0; i < 4; ++i)
    #pragma unroll
    for (int j = 0; j < 4; ++j)
      #pragma unroll
      for (int r = 0; r < 8; ++r)
        D[(size_t)(mBase + i * 16 + r + rowAdd) * N + nBase + j * 16 + l15] = acc[i][j][r];
}

// ---------------- one stochastic refinement iteration ----------------
// Block = one frame b (256 threads = 8 waves, wave c handles codebook c).
__global__ __launch_bounds__(256)
void refine_kernel(const float* __restrict__ x, const float* __restrict__ centers,
                   const float* __restrict__ fes, const float* __restrict__ xdot,
                   const float* __restrict__ G2, const float* __restrict__ censq,
                   int* __restrict__ idx, float* __restrict__ avg_probs,
                   float* __restrict__ scal, int iter, int do_stats) {
  const int b    = blockIdx.x;
  const int tid  = threadIdx.x;
  const int lane = tid & 31;
  const int c    = tid >> 5;

  __shared__ int   sIdx[CQ];
  __shared__ float sErr[DIMQ];   // x_err[d] = sum_c' centers[c',idx[c'],d] - x[b,d]
  __shared__ float sScale;

  if (tid < CQ) sIdx[tid] = idx[b * CQ + tid];
  if (tid == 0) sScale = __expf(fes[0]);
  __syncthreads();

  for (int d = tid; d < DIMQ; d += 256) {
    float s = 0.0f;
    #pragma unroll
    for (int c2 = 0; c2 < CQ; ++c2)
      s += centers[((size_t)(c2 * KQ + sIdx[c2])) * DIMQ + d];
    sErr[d] = s - x[(size_t)b * DIMQ + d];
  }
  __syncthreads();

  // a_sq[b,c] = || x_err - centers[c, idx[c]] ||^2
  const float* myCen = centers + ((size_t)(c * KQ + sIdx[c])) * DIMQ;
  float asq = 0.0f;
  for (int d = lane; d < DIMQ; d += 32) { float t = sErr[d] - myCen[d]; asq += t * t; }
  asq = wave_sum(asq);

  // cross[b,c,k] = sum_{c'!=c} G2[c'K+idx[c'], cK+k] - xdot[b, cK+k]   (lane owns k = j*32+lane)
  const float* xd = xdot + (size_t)b * NFLAT + c * KQ;
  const float* cs = censq + c * KQ;
  float cr[8];
  #pragma unroll
  for (int j = 0; j < 8; ++j) cr[j] = -xd[j * 32 + lane];
  #pragma unroll
  for (int c2 = 0; c2 < CQ; ++c2) {
    if (c2 == c) continue;
    const float* g = G2 + ((size_t)(c2 * KQ + sIdx[c2])) * NFLAT + c * KQ;
    #pragma unroll
    for (int j = 0; j < 8; ++j) cr[j] += g[j * 32 + lane];
  }

  float lg[8], sl[8];
  const float scale = sScale;
  float m = -3.0e38f;
  #pragma unroll
  for (int j = 0; j < 8; ++j) {
    lg[j] = -(asq + 2.0f * cr[j] + cs[j * 32 + lane]);
    sl[j] = lg[j] * scale;
    m = fmaxf(m, sl[j]);
  }
  m = wave_max(m);
  float se = 0.0f;
  #pragma unroll
  for (int j = 0; j < 8; ++j) se += __expf(sl[j] - m);
  se = wave_sum(se);
  const float lse = __logf(se);

  // Gumbel-max sampling on scaled logits
  float bv = -3.0e38f; int bk = 0;
  #pragma unroll
  for (int j = 0; j < 8; ++j) {
    int k = j * 32 + lane;
    unsigned int h = hash_u32(0x9E3779B9u * (unsigned)iter ^ 0x85EBCA6Bu * (unsigned)b
                              ^ 0xC2B2AE35u * (unsigned)c ^ 0x27D4EB2Fu * (unsigned)k);
    float u = (float)(h >> 9) * (1.0f / 8388608.0f) + 1.0e-10f;
    float gum = -__logf(-__logf(u));
    float v = sl[j] + gum;
    if (v > bv) { bv = v; bk = k; }
  }
  #pragma unroll
  for (int off = 16; off >= 1; off >>= 1) {
    float ov = __shfl_xor(bv, off, 32);
    int   ok = __shfl_xor(bk, off, 32);
    if (ov > bv) { bv = ov; bk = ok; }
  }
  if (lane == 0) idx[b * CQ + c] = bk;

  if (do_stats) {
    float ent = 0.0f, ess = 0.0f;
    #pragma unroll
    for (int j = 0; j < 8; ++j) {
      float lp = sl[j] - m - lse;
      float p  = __expf(lp);
      ent -= p * lp;
      ess += p * (-lg[j]);
      atomicAdd(&avg_probs[c * KQ + j * 32 + lane], p);
    }
    ent = wave_sum(ent);
    ess = wave_sum(ess);
    if (lane == 0) {
      atomicAdd(&scal[1], ent);                    // sum of per-(b,c) entropies
      atomicAdd(&scal[2], ess * (1.0f / CQ));      // sum_b mean_c expected_sumsq
    }
  }
}

// ---------------- finalize scalars ----------------
__global__ __launch_bounds__(256)
void finalize_kernel(const float* __restrict__ avg_probs, const float* __restrict__ scal,
                     float* __restrict__ out) {
  __shared__ float red[256];
  float s = 0.0f;
  for (int i = threadIdx.x; i < NFLAT; i += 256) {
    float p = avg_probs[i] * (1.0f / (float)BQ);
    s += p * __logf(p + EPSQ);
  }
  red[threadIdx.x] = s;
  __syncthreads();
  for (int off = 128; off > 0; off >>= 1) {
    if (threadIdx.x < off) red[threadIdx.x] += red[threadIdx.x + off];
    __syncthreads();
  }
  if (threadIdx.x == 0) {
    float class_entropy = -red[0] / (float)CQ;                     // mean over codebooks
    out[BQ * CQ + 0] = __logf((float)KQ) - class_entropy;          // entropy_loss
    out[BQ * CQ + 1] = scal[1] / (float)(BQ * CQ);                 // frame_entropy
    out[BQ * CQ + 2] = scal[2] / (scal[0] + EPSQ);                 // reconstruction_loss
  }
}

__global__ void copyidx_kernel(const int* __restrict__ idx, float* __restrict__ out) {
  int i = blockIdx.x * blockDim.x + threadIdx.x;
  if (i < BQ * CQ) out[i] = (float)idx[i];
}

// ---------------- host-side launch ----------------
static inline size_t align256(size_t v) { return (v + 255u) & ~(size_t)255u; }

extern "C" void kernel_launch(void* const* d_in, const int* in_sizes, int n_in,
                              void* d_out, int out_size, void* d_ws, size_t ws_size,
                              hipStream_t stream) {
  const float* x        = (const float*)d_in[0];   // (B, DIM)
  const float* centers  = (const float*)d_in[1];   // (C, K, DIM)
  const float* fes      = (const float*)d_in[2];   // (1,)
  const int*   init_idx = (const int*)  d_in[3];   // (B, C)
  // d_in[4] = num_iters (fixed at 4 in the reference)
  float* out = (float*)d_out;

  // Workspace carve-up
  char*   ws  = (char*)d_ws;
  size_t  cur = 0;
  __bf16* xb  = (__bf16*)(ws + cur); cur = align256(cur + (size_t)BQ * DIMQ * 2);
  __bf16* wb  = (__bf16*)(ws + cur); cur = align256(cur + (size_t)NFLAT * DIMQ * 2);
  float* xdot = (float*)(ws + cur);  cur = align256(cur + (size_t)BQ * NFLAT * 4);
  float* G2   = (float*)(ws + cur);  cur = align256(cur + (size_t)NFLAT * NFLAT * 4);
  float* csq  = (float*)(ws + cur);  cur = align256(cur + (size_t)NFLAT * 4);
  int*   idx  = (int*)(ws + cur);    cur = align256(cur + (size_t)BQ * CQ * 4);
  float* avg  = (float*)(ws + cur);  cur = align256(cur + (size_t)NFLAT * 4);
  float* scal = (float*)(ws + cur);  cur = align256(cur + 4 * 4);

  // 1) init accumulators + seed indexes
  init_kernel<<<(BQ * CQ + 255) / 256, 256, 0, stream>>>(init_idx, idx, avg, scal);

  // 2) bf16 conversions
  cvt_kernel<<<(BQ * DIMQ + 255) / 256, 256, 0, stream>>>(x, xb, BQ * DIMQ);
  cvt_kernel<<<(NFLAT * DIMQ + 255) / 256, 256, 0, stream>>>(centers, wb, NFLAT * DIMQ);

  // 3) ||center||^2 and sum(x*x)
  censq_kernel<<<NFLAT / 8, 256, 0, stream>>>(centers, csq);
  xsq_kernel<<<2048, 256, 0, stream>>>(x, scal, BQ * DIMQ);

  // 4) iteration-invariant WMMA GEMMs (block tile 128 x 256)
  gemm_bf16_nt<<<dim3(BQ / 128, NFLAT / 256), 256, 0, stream>>>(xb, wb, xdot, BQ, NFLAT, DIMQ);
  gemm_bf16_nt<<<dim3(NFLAT / 128, NFLAT / 256), 256, 0, stream>>>(wb, wb, G2, NFLAT, NFLAT, DIMQ);

  // 5) 4 sequential stochastic refinement iterations (stats on the last one)
  for (int it = 0; it < 4; ++it)
    refine_kernel<<<BQ, 256, 0, stream>>>(x, centers, fes, xdot, G2, csq,
                                          idx, avg, scal, it, (it == 3) ? 1 : 0);

  // 6) outputs: [indexes as float (B*C)] ++ [entropy_loss, frame_entropy, reconstruction_loss]
  copyidx_kernel<<<(BQ * CQ + 255) / 256, 256, 0, stream>>>(idx, out);
  finalize_kernel<<<1, 256, 0, stream>>>(avg, scal, out);

  (void)in_sizes; (void)n_in; (void)out_size; (void)ws_size; (void)cur;
}
